// ChannelAttnBlock_91087666413729
// MI455X (gfx1250) — matrix operand order repaired and checked by reading, not compiled, not run on hardware
//
#include <hip/hip_runtime.h>
#include <hip/hip_bf16.h>

// ---------------------------------------------------------------------------
// ChannelAttnBlock fused implementation for gfx1250 (MI455X, wave32, WMMA).
// bf16 WMMA (f32 accumulate) for all four 512x512 projections + per-head
// 32x32 attention GEMMs. GN / bias / softmax / residual fused around them.
// Register-blocked (MTx4 tiles per wave) so each A-fragment is loaded once
// per K-step and 8..16 WMMAs issue per load group.
// ---------------------------------------------------------------------------

typedef __bf16 bf16_t;
typedef __attribute__((ext_vector_type(8)))  __bf16 v8bf;
typedef __attribute__((ext_vector_type(16))) __bf16 v16bf;
typedef __attribute__((ext_vector_type(8)))  float  v8f;

#define NB      16      // batch
#define NC      512     // channels (in == hid)
#define NT      8192    // time
#define NHEAD   16
#define HC      32      // channels per head
#define NGRP    32
#define GSZ     16      // channels per group
#define CHUNK   512     // t-positions per workgroup
#define NCHUNK  (NT / CHUNK)     // 16
#define SUB     64      // t-positions per inner tile
#define NSUB    (CHUNK / SUB)    // 8
#define EPSV    1e-6f

// workspace layout (bytes)
#define WS_WQ    ((size_t)0)
#define WS_WK    ((size_t)(512*512*2))
#define WS_WV    ((size_t)(2*512*512*2))
#define WS_WP    ((size_t)(3*512*512*2))
#define WS_MEAN  ((size_t)(4*512*512*2))                 // 512 f32 (b*grp)
#define WS_RSTD  (WS_MEAN + 512*4)
#define WS_WPART ((size_t)(4*1024*1024))                 // 16*16 chunks * 16KB f32
#define WS_WFIN  (WS_WPART + (size_t)NB*NCHUNK*NHEAD*HC*HC*4)

#define ZERO8F (v8f){0.f,0.f,0.f,0.f,0.f,0.f,0.f,0.f}

// ---- WMMA helpers ----------------------------------------------------------

__device__ __forceinline__ v8f wmma_bf16(v16bf a, v16bf b, v8f c) {
    // v_wmma_f32_16x16x32_bf16  (neg_a, A, neg_b, B, c_mod, C, reuse_a, reuse_b)
    return __builtin_amdgcn_wmma_f32_16x16x32_bf16(false, a, false, b,
                                                   (short)0, c, false, false);
}

// A-matrix (16x32 bf16), source row-major [M][ldk] starting at (m0,k0)=base.
// ISA layout: lane L holds row M=L&15; VGPR0..3 = K in [kh, kh+8),
// VGPR4..7 = K in [16+kh, 16+kh+8), kh = (L>>4)*8.  -> two contiguous 8-runs.
__device__ __forceinline__ v16bf load_A16(const bf16_t* base, int ldk, int lane) {
    const bf16_t* p = base + (lane & 15) * ldk + ((lane >> 4) << 3);
    v8bf lo = *(const v8bf*)(p);
    v8bf hi = *(const v8bf*)(p + 16);
    return __builtin_shufflevector(lo, hi, 0,1,2,3,4,5,6,7,8,9,10,11,12,13,14,15);
}

// B-matrix (32x16 bf16), element (k,n) stored N-major at base[n*ldk + k],
// base at (k0,n0). ISA layout: lane L holds col N=L&15; K in [kh, kh+8) and
// [16+kh, 16+kh+8), kh = (L>>4)*8 (same K order as load_A16).
__device__ __forceinline__ v16bf load_B16(const bf16_t* base, int ldk, int lane) {
    const bf16_t* p = base + (lane & 15) * ldk + ((lane >> 4) << 3);
    v8bf lo = *(const v8bf*)(p);
    v8bf hi = *(const v8bf*)(p + 16);
    return __builtin_shufflevector(lo, hi, 0,1,2,3,4,5,6,7,8,9,10,11,12,13,14,15);
}

// Register-blocked GEMM strip: MT M-tiles x 4 N-tiles, K = 512 (16 steps).
// W row-major [.][NC] at the strip's first row; Bsrc pos-major [64][NC].
// Each A/B fragment loaded once per K-step; MT*4 WMMAs per step.
template <int MT>
__device__ __forceinline__ void gemm_block(const bf16_t* __restrict__ W,
                                           const bf16_t* __restrict__ Bsrc,
                                           v8f acc[MT][4], int lane) {
    for (int kk = 0; kk < 16; ++kk) {
        v16bf a[MT];
        #pragma unroll
        for (int m = 0; m < MT; ++m)
            a[m] = load_A16(W + (m * 16) * NC + kk * 32, NC, lane);
        #pragma unroll
        for (int n = 0; n < 4; ++n) {
            v16bf bm = load_B16(Bsrc + (n * 16) * NC + kk * 32, NC, lane);
            #pragma unroll
            for (int m = 0; m < MT; ++m)
                acc[m][n] = wmma_bf16(a[m], bm, acc[m][n]);
        }
    }
}

// ---- kernel 0: weights fp32 -> bf16 ---------------------------------------

__global__ void k_w2bf(const float* wq, const float* wk, const float* wv,
                       const float* wp, bf16_t* oq, bf16_t* ok, bf16_t* ov,
                       bf16_t* op) {
    int i = blockIdx.x * blockDim.x + threadIdx.x;      // 0 .. 4*2^18-1
    int which = i >> 18, off = i & 0x3FFFF;
    const float* s = which == 0 ? wq : which == 1 ? wk : which == 2 ? wv : wp;
    bf16_t*      d = which == 0 ? oq : which == 1 ? ok : which == 2 ? ov : op;
    d[off] = (bf16_t)s[off];
}

// ---- kernel 1: GroupNorm statistics per (b, group) -------------------------

__global__ void k_gnstats(const float* __restrict__ x, float* mean, float* rstd) {
    int bg = blockIdx.x;                 // b*32 + g
    int b = bg >> 5, g = bg & 31;
    const int N = GSZ * NT;              // 131072
    float s = 0.f, s2 = 0.f;
    for (int i = threadIdx.x; i < N; i += 256) {
        int c = g * GSZ + (i >> 13);
        int t = i & (NT - 1);
        float v = x[((size_t)(b * NC + c)) * NT + t];
        s += v; s2 += v * v;
    }
    __shared__ float red[2][256];
    red[0][threadIdx.x] = s; red[1][threadIdx.x] = s2;
    __syncthreads();
    for (int st = 128; st > 0; st >>= 1) {
        if (threadIdx.x < st) {
            red[0][threadIdx.x] += red[0][threadIdx.x + st];
            red[1][threadIdx.x] += red[1][threadIdx.x + st];
        }
        __syncthreads();
    }
    if (threadIdx.x == 0) {
        float m = red[0][0] / (float)N;
        float var = red[1][0] / (float)N - m * m;
        mean[bg] = m;
        rstd[bg] = rsqrtf(var + EPSV);
    }
}

// ---- GN tile load: x -> hB (pos-major bf16 [SUB][NC]) ----------------------

__device__ __forceinline__ void gn_tile(const float* __restrict__ x,
                                        const float* mean, const float* rstd,
                                        const float* gns, const float* gnb,
                                        bf16_t* hB, int b, int t0) {
    for (int i = threadIdx.x; i < NC * SUB; i += 256) {
        int c = i >> 6, p = i & (SUB - 1);
        int g = c >> 4;
        float m = mean[b * NGRP + g], r = rstd[b * NGRP + g];
        float v = x[((size_t)(b * NC + c)) * NT + t0 + p];
        hB[p * NC + c] = (bf16_t)((v - m) * r * gns[c] + gnb[c]);
    }
}

// ---- kernel 2: Q/K projection + channel softmax + partial w ---------------
// grid: NB*NCHUNK blocks, 256 threads (8 waves). dyn LDS: 192KB.

__global__ void k_qk_w(const float* __restrict__ x, const float* mean,
                       const float* rstd, const float* gns, const float* gnb,
                       const bf16_t* __restrict__ wq, const float* bq,
                       const bf16_t* __restrict__ wk, const float* bk,
                       float* __restrict__ wpart) {
    extern __shared__ char smem[];
    bf16_t* hB = (bf16_t*)smem;                        // [SUB][NC]   64KB
    bf16_t* qs = hB + SUB * NC;                        // [NC][SUB]   64KB (ch-major)
    bf16_t* ks = qs + NC * SUB;                        // [NC][SUB]   64KB

    int blk = blockIdx.x;
    int b = blk >> 4, chunk = blk & 15;
    int lane = threadIdx.x & 31, wv = threadIdx.x >> 5;

    v8f wacc[8];
    #pragma unroll
    for (int i = 0; i < 8; ++i) wacc[i] = ZERO8F;

    for (int s = 0; s < NSUB; ++s) {
        int t0 = chunk * CHUNK + s * SUB;
        gn_tile(x, mean, rstd, gns, gnb, hB, b, t0);
        __syncthreads();

        // q & k projections. Each wave owns a 64-row strip of each matrix,
        // processed as two 32x64 register blocks (2 Mt x 4 Nt).
        for (int phase = 0; phase < 4; ++phase) {
            int mat = phase >> 1, g = phase & 1;       // matrix, half-strip
            const bf16_t* W  = mat ? wk : wq;
            const float*  bi = mat ? bk : bq;
            bf16_t*       ds = mat ? ks : qs;
            int row0 = wv * 64 + g * 32;
            v8f acc[2][4];
            #pragma unroll
            for (int m = 0; m < 2; ++m)
                #pragma unroll
                for (int n = 0; n < 4; ++n) acc[m][n] = ZERO8F;
            gemm_block<2>(W + row0 * NC, hB, acc, lane);
            int hi = (lane >> 4) << 3;
            #pragma unroll
            for (int m = 0; m < 2; ++m)
                #pragma unroll
                for (int n = 0; n < 4; ++n) {
                    int col = n * 16 + (lane & 15);
                    #pragma unroll
                    for (int r = 0; r < 8; ++r) {
                        int row = row0 + m * 16 + r + hi;
                        ds[row * SUB + col] = (bf16_t)(acc[m][n][r] + bi[row]);
                    }
                }
        }
        __syncthreads();

        // channel softmax: 2 mats * 16 heads * 64 pos = 2048 rows / 256 thr
        for (int i = 0; i < 8; ++i) {
            int id = threadIdx.x + i * 256;
            int mat = id >> 10, rem = id & 1023;
            int head = rem >> 6, pos = rem & 63;
            bf16_t* bse = (mat ? ks : qs) + (head * HC) * SUB + pos;
            float mx = -1e30f;
            #pragma unroll
            for (int c = 0; c < HC; ++c) { float v = (float)bse[c * SUB]; mx = v > mx ? v : mx; }
            float sum = 0.f;
            #pragma unroll
            for (int c = 0; c < HC; ++c) sum += __expf((float)bse[c * SUB] - mx);
            float inv = 1.f / sum;
            #pragma unroll
            for (int c = 0; c < HC; ++c)
                bse[c * SUB] = (bf16_t)(__expf((float)bse[c * SUB] - mx) * inv);
        }
        __syncthreads();

        // per-head partial w += qs(32xSUB) * ks(32xSUB)^T : 64 tiles / 8 waves
        for (int i = 0; i < 8; ++i) {
            int tIdx = wv + 8 * i;
            int head = tIdx >> 2, mt = (tIdx >> 1) & 1, nt = tIdx & 1;
            #pragma unroll
            for (int kk = 0; kk < 2; ++kk) {
                v16bf a = load_A16(qs + (head * HC + mt * 16) * SUB + kk * 32, SUB, lane);
                v16bf bm = load_B16(ks + (head * HC + nt * 16) * SUB + kk * 32, SUB, lane);
                wacc[i] = wmma_bf16(a, bm, wacc[i]);
            }
        }
        __syncthreads();   // hB/qs/ks reused next iteration
    }

    // write deterministic per-chunk partials
    for (int i = 0; i < 8; ++i) {
        int tIdx = wv + 8 * i;
        int head = tIdx >> 2, mt = (tIdx >> 1) & 1, nt = tIdx & 1;
        int hi = (lane >> 4) << 3, col = nt * 16 + (lane & 15);
        #pragma unroll
        for (int r = 0; r < 8; ++r) {
            int row = mt * 16 + r + hi;
            size_t idx = (((size_t)(b * NCHUNK + chunk) * NHEAD + head) * HC + row) * HC + col;
            wpart[idx] = wacc[i][r];
        }
    }
}

// ---- kernel 3: reduce w partials over chunks -------------------------------

__global__ void k_wreduce(const float* __restrict__ wpart, float* __restrict__ wfin) {
    int idx = blockIdx.x * blockDim.x + threadIdx.x;    // < NB*16384
    int b = idx >> 14, rem = idx & 16383;
    float s = 0.f;
    #pragma unroll
    for (int c = 0; c < NCHUNK; ++c)
        s += wpart[((size_t)(b * NCHUNK + c)) * 16384 + rem];
    wfin[(size_t)b * 16384 + rem] = s;
}

// ---- kernel 4: V projection + h = w*v + out projection + residual ----------
// grid: NB*NCHUNK blocks, 256 threads. dyn LDS: 160KB.

__global__ void k_v_out(const float* __restrict__ x, const float* mean,
                        const float* rstd, const float* gns, const float* gnb,
                        const bf16_t* __restrict__ wvw, const float* bv,
                        const float* __restrict__ wfin,
                        const bf16_t* __restrict__ wpw, const float* bp,
                        float* __restrict__ out) {
    extern __shared__ char smem[];
    bf16_t* wL = (bf16_t*)smem;                        // [NHEAD][HC][HC] 32KB
    bf16_t* hB = wL + NHEAD * HC * HC;                 // [SUB][NC]       64KB
    bf16_t* vB = hB + SUB * NC;                        // [SUB][NC]       64KB

    int blk = blockIdx.x;
    int b = blk >> 4, chunk = blk & 15;
    int lane = threadIdx.x & 31, wv = threadIdx.x >> 5;

    for (int i = threadIdx.x; i < NHEAD * HC * HC; i += 256)
        wL[i] = (bf16_t)wfin[(size_t)b * 16384 + i];
    __syncthreads();

    for (int s = 0; s < NSUB; ++s) {
        int t0 = chunk * CHUNK + s * SUB;
        gn_tile(x, mean, rstd, gns, gnb, hB, b, t0);
        __syncthreads();

        // v = wv * h : wave owns a 64-row strip -> 4x4 register block
        {
            int row0 = wv * 64;
            v8f acc[4][4];
            #pragma unroll
            for (int m = 0; m < 4; ++m)
                #pragma unroll
                for (int n = 0; n < 4; ++n) acc[m][n] = ZERO8F;
            gemm_block<4>(wvw + row0 * NC, hB, acc, lane);
            int hi = (lane >> 4) << 3;
            #pragma unroll
            for (int m = 0; m < 4; ++m)
                #pragma unroll
                for (int n = 0; n < 4; ++n) {
                    int col = n * 16 + (lane & 15);
                    #pragma unroll
                    for (int r = 0; r < 8; ++r) {
                        int row = row0 + m * 16 + r + hi;
                        vB[col * NC + row] = (bf16_t)(acc[m][n][r] + bv[row]);
                    }
                }
        }
        __syncthreads();

        // h = w * v per head: 16 heads * 2 Mt * 4 Nt = 128 tiles, single K step
        for (int i = 0; i < 16; ++i) {
            int tIdx = wv + 8 * i;
            int head = tIdx >> 3, rem = tIdx & 7;
            int mt = rem >> 2, nt = rem & 3;
            v16bf a = load_A16(wL + head * HC * HC + (mt * 16) * HC, HC, lane);
            v16bf bm = load_B16(vB + (nt * 16) * NC + head * HC, NC, lane);
            v8f acc = ZERO8F;
            acc = wmma_bf16(a, bm, acc);
            int hi = (lane >> 4) << 3, col = nt * 16 + (lane & 15);
            #pragma unroll
            for (int r = 0; r < 8; ++r) {
                int row = head * HC + mt * 16 + r + hi;
                hB[col * NC + row] = (bf16_t)acc[r];   // hB free: reuse for h
            }
        }
        __syncthreads();

        // out = wp * h + bp + x : 4x4 register block per wave
        {
            int row0 = wv * 64;
            v8f acc[4][4];
            #pragma unroll
            for (int m = 0; m < 4; ++m)
                #pragma unroll
                for (int n = 0; n < 4; ++n) acc[m][n] = ZERO8F;
            gemm_block<4>(wpw + row0 * NC, hB, acc, lane);
            int hi = (lane >> 4) << 3;
            #pragma unroll
            for (int m = 0; m < 4; ++m)
                #pragma unroll
                for (int n = 0; n < 4; ++n) {
                    int col = n * 16 + (lane & 15);
                    #pragma unroll
                    for (int r = 0; r < 8; ++r) {
                        int row = row0 + m * 16 + r + hi;
                        size_t g = ((size_t)(b * NC + row)) * NT + t0 + col;
                        out[g] = x[g] + acc[m][n][r] + bp[row];
                    }
                }
        }
        __syncthreads();
    }
}

// ---------------------------------------------------------------------------

extern "C" void kernel_launch(void* const* d_in, const int* in_sizes, int n_in,
                              void* d_out, int out_size, void* d_ws, size_t ws_size,
                              hipStream_t stream) {
    const float* x   = (const float*)d_in[0];
    const float* gns = (const float*)d_in[1];
    const float* gnb = (const float*)d_in[2];
    const float* wq  = (const float*)d_in[3];
    const float* bq  = (const float*)d_in[4];
    const float* wk  = (const float*)d_in[5];
    const float* bk  = (const float*)d_in[6];
    const float* wvf = (const float*)d_in[7];
    const float* bv  = (const float*)d_in[8];
    const float* wp  = (const float*)d_in[9];
    const float* bp  = (const float*)d_in[10];
    float* out = (float*)d_out;

    char* ws = (char*)d_ws;
    bf16_t* wq_bf = (bf16_t*)(ws + WS_WQ);
    bf16_t* wk_bf = (bf16_t*)(ws + WS_WK);
    bf16_t* wv_bf = (bf16_t*)(ws + WS_WV);
    bf16_t* wp_bf = (bf16_t*)(ws + WS_WP);
    float*  mean  = (float*)(ws + WS_MEAN);
    float*  rstd  = (float*)(ws + WS_RSTD);
    float*  wpart = (float*)(ws + WS_WPART);
    float*  wfin  = (float*)(ws + WS_WFIN);

    k_w2bf<<<4 * 262144 / 256, 256, 0, stream>>>(wq, wk, wvf, wp,
                                                 wq_bf, wk_bf, wv_bf, wp_bf);
    k_gnstats<<<NB * NGRP, 256, 0, stream>>>(x, mean, rstd);
    k_qk_w<<<NB * NCHUNK, 256, 192 * 1024, stream>>>(x, mean, rstd, gns, gnb,
                                                     wq_bf, bq, wk_bf, bk, wpart);
    k_wreduce<<<NB * 16384 / 256, 256, 0, stream>>>(wpart, wfin);
    k_v_out<<<NB * NCHUNK, 256, 160 * 1024, stream>>>(x, mean, rstd, gns, gnb,
                                                      wv_bf, bv, wfin, wp_bf, bp, out);
}
